// Attention_62156766707792
// MI455X (gfx1250) — matrix operand order repaired and checked by reading, not compiled, not run on hardware
//
#include <hip/hip_runtime.h>
#include <hip/hip_bf16.h>

typedef _Float16 v16h __attribute__((ext_vector_type(16)));
typedef _Float16 v8h  __attribute__((ext_vector_type(8)));
typedef float    v8f  __attribute__((ext_vector_type(8)));

#define Bq   4
#define Nq   2048
#define Cq   1024
#define Hq   16
#define Dq   64
#define SCALEq 0.125f   // 64^-0.5

static __device__ __forceinline__ v16h make_frag(const _Float16* lo, const _Float16* hi) {
    v8h a = *(const v8h*)lo;
    v8h b = *(const v8h*)hi;
    v16h r;
#pragma unroll
    for (int i = 0; i < 8; ++i) { r[i] = a[i]; r[i + 8] = b[i]; }
    return r;
}

// ---------------------------------------------------------------------------
// fp32 -> f16 conversion, 8 elements / thread (b128-friendly)
// ---------------------------------------------------------------------------
__global__ void f32_to_f16_kernel(const float* __restrict__ src,
                                  _Float16* __restrict__ dst, int n) {
    int i = blockIdx.x * blockDim.x + threadIdx.x;
    int stride = gridDim.x * blockDim.x;
    int nch = n >> 3;
    for (; i < nch; i += stride) {
        const float* s = src + (size_t)i * 8;
        v8h d;
#pragma unroll
        for (int j = 0; j < 8; ++j) d[j] = (_Float16)s[j];
        *(v8h*)&dst[(size_t)i * 8] = d;
    }
}

// ---------------------------------------------------------------------------
// WMMA f16 GEMM: out[m,n] = sum_k A[m,k] * W[n,k] + bias[n]
// BM=128, BN=128, BK=32; double-buffered LDS, 1 barrier / K-step.
// 8 waves: 2(M) x 4(N); wave tile 64x32 = 4x2 WMMA frags.
// mode 0: scatter qkv (q,k row-major [B,H,N,D]; v transposed [B,H,D,N]), f16
// mode 1: fp32 output [M, 1024] + bias
// ---------------------------------------------------------------------------
#define BM 128
#define BN 128
#define BK 32
#define LDT 40   // BK + 8 pad (80 B rows, 16B aligned)

__global__ __launch_bounds__(256)
void gemm_wmma_f16(const _Float16* __restrict__ A,
                   const _Float16* __restrict__ W,
                   const float* __restrict__ bias,
                   int M, int Nout, int K, int mode,
                   _Float16* __restrict__ qb,
                   _Float16* __restrict__ kb,
                   _Float16* __restrict__ vtb,
                   float* __restrict__ outf) {
    __shared__ __align__(16) _Float16 As[2][BM][LDT];
    __shared__ __align__(16) _Float16 Bs[2][BN][LDT];

    const int tid  = threadIdx.x;
    const int wave = tid >> 5;
    const int lane = tid & 31;
    const int l16  = lane & 15;
    const int lhi  = lane >> 4;
    const int m0blk = blockIdx.y * BM;
    const int n0blk = blockIdx.x * BN;
    const int wm = wave >> 2;   // 0..1  (64 rows each)
    const int wn = wave & 3;    // 0..3  (32 cols each)

    // cooperative-load decomposition (shared by A and B tiles: 512 b128 chunks)
    const int lr0 = tid >> 2;            // rows 0..63  (+64 for second chunk)
    const int lc  = (tid & 3) * 8;       // k-offset 0,8,16,24

    v8f acc[4][2];
    {
        v8f z = {};
#pragma unroll
        for (int i = 0; i < 4; ++i)
#pragma unroll
            for (int j = 0; j < 2; ++j) acc[i][j] = z;
    }

    // prologue: tile 0 straight into buffer 0
#pragma unroll
    for (int i = 0; i < 2; ++i) {
        int r = lr0 + i * 64;
        *(v8h*)&As[0][r][lc] = *(const v8h*)&A[(size_t)(m0blk + r) * K + lc];
        *(v8h*)&Bs[0][r][lc] = *(const v8h*)&W[(size_t)(n0blk + r) * K + lc];
    }
    __syncthreads();

    const int nk = K / BK;
    for (int kk = 0; kk < nk; ++kk) {
        const int cur = kk & 1, nxt = cur ^ 1;
        const bool hasNext = (kk + 1) < nk;
        v8h ta[2], tb[2];
        if (hasNext) {
            const int k0n = (kk + 1) * BK;
#pragma unroll
            for (int i = 0; i < 2; ++i) {
                int r = lr0 + i * 64;
                ta[i] = *(const v8h*)&A[(size_t)(m0blk + r) * K + k0n + lc];
                tb[i] = *(const v8h*)&W[(size_t)(n0blk + r) * K + k0n + lc];
            }
            if (kk + 2 < nk) {   // global_prefetch_b8 for the tile after next
                const int k0p = (kk + 2) * BK;
                __builtin_prefetch(&A[(size_t)(m0blk + lr0) * K + k0p], 0, 1);
                __builtin_prefetch(&W[(size_t)(n0blk + lr0) * K + k0p], 0, 1);
            }
        }

        // compute on current buffer
        v16h af[4], bf[2];
#pragma unroll
        for (int i = 0; i < 4; ++i) {
            int row = wm * 64 + i * 16 + l16;
            af[i] = make_frag(&As[cur][row][lhi * 8], &As[cur][row][16 + lhi * 8]);
        }
#pragma unroll
        for (int j = 0; j < 2; ++j) {
            int row = wn * 32 + j * 16 + l16;
            bf[j] = make_frag(&Bs[cur][row][lhi * 8], &Bs[cur][row][16 + lhi * 8]);
        }
#pragma unroll
        for (int i = 0; i < 4; ++i)
#pragma unroll
            for (int j = 0; j < 2; ++j)
                acc[i][j] = __builtin_amdgcn_wmma_f32_16x16x32_f16(
                    false, af[i], false, bf[j], (short)0, acc[i][j], false, false);

        if (hasNext) {
#pragma unroll
            for (int i = 0; i < 2; ++i) {
                int r = lr0 + i * 64;
                *(v8h*)&As[nxt][r][lc] = ta[i];
                *(v8h*)&Bs[nxt][r][lc] = tb[i];
            }
        }
        __syncthreads();
    }

    // epilogue
#pragma unroll
    for (int i = 0; i < 4; ++i) {
#pragma unroll
        for (int j = 0; j < 2; ++j) {
            int colb = n0blk + wn * 32 + j * 16 + l16;
            float bv = bias[colb];
#pragma unroll
            for (int r = 0; r < 8; ++r) {
                int row = m0blk + wm * 64 + i * 16 + r + 8 * lhi;
                float val = acc[i][j][r] + bv;
                if (mode == 0) {
                    int region = colb >> 10;          // 0=q 1=k 2=v
                    int h = (colb & (Cq - 1)) >> 6;   // /64
                    int d = colb & (Dq - 1);
                    int b = row >> 11;                // /2048
                    int n = row & (Nq - 1);
                    size_t bh = (size_t)(b * Hq + h);
                    if (region == 0)
                        qb[(bh * Nq + n) * Dq + d] = (_Float16)val;
                    else if (region == 1)
                        kb[(bh * Nq + n) * Dq + d] = (_Float16)val;
                    else
                        vtb[(bh * Dq + d) * Nq + n] = (_Float16)val;
                } else {
                    outf[(size_t)row * Cq + colb] = val;
                }
            }
        }
    }
}

// ---------------------------------------------------------------------------
// Flash attention: workgroup = 128 query rows of one (b,h); 8 waves x 16 rows.
// Key tiles of 64, double-buffered K/V in LDS, online softmax, P staged in
// per-wave LDS (wave-local RAW -> s_wait_dscnt, no cross-wave barrier).
// ---------------------------------------------------------------------------
#define TK 64
#define LDK 72   // 64 + 8 pad

__global__ __launch_bounds__(256)
void flash_attn_wmma(const _Float16* __restrict__ q,
                     const _Float16* __restrict__ k,
                     const _Float16* __restrict__ vt,
                     _Float16* __restrict__ aout) {  // [B*N, C] f16
    __shared__ __align__(16) _Float16 Ks[2][TK][LDK];     // [key][d]
    __shared__ __align__(16) _Float16 Vs[2][Dq][LDK];     // [d][key]
    __shared__ __align__(16) _Float16 Ps[8][16][LDK];     // per-wave P tile

    const int tid  = threadIdx.x;
    const int wave = tid >> 5;
    const int lane = tid & 31;
    const int l16  = lane & 15;
    const int lhi  = lane >> 4;
    const int bh = blockIdx.y;       // 0..63
    const int b  = bh >> 4;
    const int h  = bh & 15;
    const int q0 = blockIdx.x * 128 + wave * 16;

    const _Float16* qbase = q  + ((size_t)bh * Nq + q0) * Dq;
    const _Float16* kbase = k  + (size_t)bh * Nq * Dq;
    const _Float16* vbase = vt + (size_t)bh * Dq * Nq;

    // Q fragments (d-chunks {0..31, 32..63}), 1/sqrt(d) folded in (exact: 2^-3)
    v16h qf[2];
#pragma unroll
    for (int t = 0; t < 2; ++t) {
        const _Float16* rp = qbase + (size_t)l16 * Dq + t * 32;
        qf[t] = make_frag(rp + lhi * 8, rp + 16 + lhi * 8);
#pragma unroll
        for (int e = 0; e < 16; ++e) qf[t][e] = qf[t][e] * (_Float16)SCALEq;
    }

    v8f acc[4];
    {
        v8f z = {};
#pragma unroll
        for (int t = 0; t < 4; ++t) acc[t] = z;
    }
    float mrow[8], lrow[8];
#pragma unroll
    for (int r = 0; r < 8; ++r) { mrow[r] = -1e30f; lrow[r] = 0.0f; }

    // cooperative-load decomposition: 512 b128 chunks per 64x64 tile
    const int lr0 = tid >> 3;         // rows 0..31 (+32)
    const int lc  = (tid & 7) * 8;    // col offset 0..56

    // prologue: tile 0 into buffer 0
#pragma unroll
    for (int i = 0; i < 2; ++i) {
        int r = lr0 + i * 32;
        *(v8h*)&Ks[0][r][lc] = *(const v8h*)&kbase[(size_t)r * Dq + lc];
        *(v8h*)&Vs[0][r][lc] = *(const v8h*)&vbase[(size_t)r * Nq + lc];
    }
    __syncthreads();

    const int nit = Nq / TK;   // 32
    for (int it = 0; it < nit; ++it) {
        const int cur = it & 1, nxt = cur ^ 1;
        const bool hasNext = (it + 1) < nit;
        v8h tk[2], tv[2];
        if (hasNext) {
            const int ktn = (it + 1) * TK;
#pragma unroll
            for (int i = 0; i < 2; ++i) {
                int r = lr0 + i * 32;
                tk[i] = *(const v8h*)&kbase[(size_t)(ktn + r) * Dq + lc];
                tv[i] = *(const v8h*)&vbase[(size_t)r * Nq + ktn + lc];
            }
        }

        // S = (Q*scale) K^T : 4 key blocks x 2 d-chunks
        v8f s[4];
        {
            v8f z = {};
#pragma unroll
            for (int jb = 0; jb < 4; ++jb) s[jb] = z;
        }
#pragma unroll
        for (int t = 0; t < 2; ++t) {
#pragma unroll
            for (int jb = 0; jb < 4; ++jb) {
                v16h bfr = make_frag(&Ks[cur][jb * 16 + l16][t * 32 + lhi * 8],
                                     &Ks[cur][jb * 16 + l16][t * 32 + 16 + lhi * 8]);
                s[jb] = __builtin_amdgcn_wmma_f32_16x16x32_f16(
                    false, qf[t], false, bfr, (short)0, s[jb], false, false);
            }
        }

        // online softmax; row = r + 8*lhi per lane, 16 columns across lanes
#pragma unroll
        for (int r = 0; r < 8; ++r) {
            float a0 = s[0][r], a1 = s[1][r], a2 = s[2][r], a3 = s[3][r];
            float mx = fmaxf(fmaxf(a0, a1), fmaxf(a2, a3));
            mx = fmaxf(mx, __shfl_xor(mx, 1, 32));
            mx = fmaxf(mx, __shfl_xor(mx, 2, 32));
            mx = fmaxf(mx, __shfl_xor(mx, 4, 32));
            mx = fmaxf(mx, __shfl_xor(mx, 8, 32));
            float mnew  = fmaxf(mrow[r], mx);
            float alpha = __expf(mrow[r] - mnew);
            mrow[r] = mnew;
            float p0 = __expf(a0 - mnew);
            float p1 = __expf(a1 - mnew);
            float p2 = __expf(a2 - mnew);
            float p3 = __expf(a3 - mnew);
            float rs = (p0 + p1) + (p2 + p3);
            rs += __shfl_xor(rs, 1, 32);
            rs += __shfl_xor(rs, 2, 32);
            rs += __shfl_xor(rs, 4, 32);
            rs += __shfl_xor(rs, 8, 32);
            lrow[r] = lrow[r] * alpha + rs;
#pragma unroll
            for (int t = 0; t < 4; ++t) acc[t][r] *= alpha;
            int row = r + 8 * lhi;
            Ps[wave][row][l16]      = (_Float16)p0;
            Ps[wave][row][16 + l16] = (_Float16)p1;
            Ps[wave][row][32 + l16] = (_Float16)p2;
            Ps[wave][row][48 + l16] = (_Float16)p3;
        }
        // wave-local LDS RAW: DS ops are in-order within a wave; just fence
        // the compiler and drain DScnt — no cross-wave barrier needed.
        asm volatile("s_wait_dscnt 0x0" ::: "memory");

        // O += P V : A = P[16x64] (2 k-chunks), B = V[32 x 16d] per d-block
        v16h pf[2];
#pragma unroll
        for (int t = 0; t < 2; ++t)
            pf[t] = make_frag(&Ps[wave][l16][t * 32 + lhi * 8],
                              &Ps[wave][l16][t * 32 + 16 + lhi * 8]);
#pragma unroll
        for (int t = 0; t < 2; ++t) {
#pragma unroll
            for (int db = 0; db < 4; ++db) {
                v16h bfr = make_frag(&Vs[cur][db * 16 + l16][t * 32 + lhi * 8],
                                     &Vs[cur][db * 16 + l16][t * 32 + 16 + lhi * 8]);
                acc[db] = __builtin_amdgcn_wmma_f32_16x16x32_f16(
                    false, pf[t], false, bfr, (short)0, acc[db], false, false);
            }
        }

        if (hasNext) {
#pragma unroll
            for (int i = 0; i < 2; ++i) {
                int r = lr0 + i * 32;
                *(v8h*)&Ks[nxt][r][lc] = tk[i];
                *(v8h*)&Vs[nxt][r][lc] = tv[i];
            }
        }
        __syncthreads();
    }

    // normalize + write f16 [B*N, C] at column h*64
#pragma unroll
    for (int r = 0; r < 8; ++r) {
        int row = r + 8 * lhi;
        float rinv = 1.0f / lrow[r];
        size_t base = ((size_t)b * Nq + q0 + row) * Cq + h * Dq;
#pragma unroll
        for (int t = 0; t < 4; ++t)
            aout[base + t * 16 + l16] = (_Float16)(acc[t][r] * rinv);
    }
}

// ---------------------------------------------------------------------------
extern "C" void kernel_launch(void* const* d_in, const int* in_sizes, int n_in,
                              void* d_out, int out_size, void* d_ws, size_t ws_size,
                              hipStream_t stream) {
    const float* x      = (const float*)d_in[0];
    const float* w_qkv  = (const float*)d_in[1];
    const float* b_qkv  = (const float*)d_in[2];
    const float* w_proj = (const float*)d_in[3];
    const float* b_proj = (const float*)d_in[4];
    float* out = (float*)d_out;

    const int M  = Bq * Nq;        // 8192
    const int K  = Cq;             // 1024
    const int N3 = 3 * Cq;         // 3072

    char* ws = (char*)d_ws;
    size_t off = 0;
    _Float16* xh     = (_Float16*)(ws + off); off += (size_t)M * K * 2;
    _Float16* wqkvh  = (_Float16*)(ws + off); off += (size_t)N3 * K * 2;
    _Float16* wprojh = (_Float16*)(ws + off); off += (size_t)Cq * K * 2;
    _Float16* qb     = (_Float16*)(ws + off); off += (size_t)M * Cq * 2;
    _Float16* kb     = (_Float16*)(ws + off); off += (size_t)M * Cq * 2;
    _Float16* vtb    = (_Float16*)(ws + off); off += (size_t)M * Cq * 2;
    _Float16* aoh    = (_Float16*)(ws + off); off += (size_t)M * Cq * 2;

    // 1) fp32 -> f16 conversions
    f32_to_f16_kernel<<<1024, 256, 0, stream>>>(x, xh, M * K);
    f32_to_f16_kernel<<<1024, 256, 0, stream>>>(w_qkv, wqkvh, N3 * K);
    f32_to_f16_kernel<<<512, 256, 0, stream>>>(w_proj, wprojh, Cq * K);

    // 2) QKV projection + scatter (v transposed)
    gemm_wmma_f16<<<dim3(N3 / BN, M / BM), 256, 0, stream>>>(
        xh, wqkvh, b_qkv, M, N3, K, /*mode=*/0, qb, kb, vtb, nullptr);

    // 3) flash attention
    flash_attn_wmma<<<dim3(Nq / 128, Bq * Hq), 256, 0, stream>>>(qb, kb, vtb, aoh);

    // 4) output projection (fp32 out + bias)
    gemm_wmma_f16<<<dim3(Cq / BN, M / BM), 256, 0, stream>>>(
        aoh, wprojh, b_proj, M, Cq, K, /*mode=*/1, nullptr, nullptr, nullptr, out);
}
